// MultiModalEncoder_51831665328266
// MI455X (gfx1250) — matrix operand order repaired and checked by reading, not compiled, not run on hardware
//
#include <hip/hip_runtime.h>

typedef __attribute__((ext_vector_type(16))) _Float16 v16h;
typedef __attribute__((ext_vector_type(8)))  _Float16 v8h;
typedef __attribute__((ext_vector_type(8)))  float    v8f;

#define SRC_N   512
#define FEAT    256
#define OUTC    200

// workspace layout (in _Float16 elements) -- all 32B aligned
#define OFF_W1T   0        // sg_w1^T   [256][256]
#define OFF_W2T   65536    // sg_w2^T   [256][256]
#define OFF_SEWT  131072   // se_w^T    [112][256] (rows 100..111 zero)
#define OFF_PW2T  159744   // p_w2^T    [128][64]
#define OFF_PW3T  167936   // p_w3^T    [256][128]
#define OFF_OEWT  200704   // oe_w^T    [100][256]

// ---------------------------------------------------------------------------
// weight convert + transpose: src [K][N] f32 -> dst [Nd][K] f16 (pad rows = 0)
// ---------------------------------------------------------------------------
__global__ void cvt_transpose(const float* __restrict__ src,
                              _Float16* __restrict__ dst,
                              int K, int N, int Nd) {
  int idx = blockIdx.x * blockDim.x + threadIdx.x;
  int total = Nd * K;
  if (idx >= total) return;
  int n = idx / K, k = idx - n * K;
  float v = (n < N) ? src[k * N + n] : 0.0f;
  dst[idx] = (_Float16)v;
}

// ---------------------------------------------------------------------------
// WMMA fragment helpers (wave32, 16x16x32 f16 -> f32)
// A (16x32, 16-bit): lane L: row = m + (L&15); k in [h*16, h*16+16),
//                    h = L>>4. Two adjacent 16B chunks.
// B (32x16, 16-bit): lane L: col = n + (L&15); k in [h*16, h*16+16).
//                    One 32B contiguous chunk from transposed [N][K] weights.
// C (16x16 f32):     lane L: col n + (L&15); VGPR j -> row m + j + 8*(L>>4).
// ---------------------------------------------------------------------------
__device__ __forceinline__ v16h load_a_lds(const _Float16* base, int ldk,
                                           int mbase, int kbase, int lane) {
  int hh = lane >> 4;
  const _Float16* p = base + (mbase + (lane & 15)) * ldk + kbase + hh * 16;
  v8h lo = *(const v8h*)p;
  v8h hi = *(const v8h*)(p + 8);
  v16h a;
#pragma unroll
  for (int i = 0; i < 8; ++i) { a[i] = lo[i]; a[i + 8] = hi[i]; }
  return a;
}

__device__ __forceinline__ v16h load_b_glb(const _Float16* wt, int ldk,
                                           int nbase, int kbase, int lane) {
  int hh = lane >> 4;
  const _Float16* p = wt + (size_t)(nbase + (lane & 15)) * ldk + kbase + hh * 16;
  return *(const v16h*)p;
}

__device__ __forceinline__ v8f wmma16(v16h a, v16h b, v8f c) {
  return __builtin_amdgcn_wmma_f32_16x16x32_f16(false, a, false, b,
                                                (short)0, c, false, false);
}

// ---------------------------------------------------------------------------
// SG path: out[:, 0:100] = (relu(X@W1+b1)@W2+b2) @ se_w + se_b
// X rows gathered: row r -> batch r>>10, local r&1023 (src if local<512)
// 256 blocks x 64 rows, 256 threads (8 waves)
// ---------------------------------------------------------------------------
__global__ __launch_bounds__(256) void sg_kernel(
    const float* __restrict__ srcf, const float* __restrict__ reff,
    const _Float16* __restrict__ w1t, const _Float16* __restrict__ w2t,
    const _Float16* __restrict__ sewt,
    const float* __restrict__ b1, const float* __restrict__ b2,
    const float* __restrict__ seb, float* __restrict__ out) {
  __shared__ __align__(16) _Float16 X[64 * 256];
  __shared__ __align__(16) _Float16 A1[64 * 256];

  const int t = threadIdx.x;
  const int lane = t & 31, w = t >> 5;
  const int rbase = blockIdx.x * 64;

  // ---- gather + f32->f16 into LDS: thread owns (row t>>2, 64-col quarter)
  {
    int rr = t >> 2, q = t & 3;
    int r = rbase + rr;
    int b = r >> 10;
    int local = r & 1023;
    const float* rowp = (local < SRC_N)
        ? (srcf + (size_t)(b * SRC_N + local) * FEAT)
        : (reff + (size_t)(b * SRC_N + (local - SRC_N)) * FEAT);
    const float4* r4 = (const float4*)rowp + q * 16;
    _Float16* xr = X + rr * 256 + q * 64;
#pragma unroll
    for (int i = 0; i < 16; ++i) {
      float4 v = r4[i];
      xr[i * 4 + 0] = (_Float16)v.x; xr[i * 4 + 1] = (_Float16)v.y;
      xr[i * 4 + 2] = (_Float16)v.z; xr[i * 4 + 3] = (_Float16)v.w;
    }
  }
  __syncthreads();

  const int n0 = (2 * w) * 16, n1 = (2 * w + 1) * 16;

  // ---- GEMM1: A1 = relu(X @ W1 + b1)   (M=64,N=256,K=256)
  {
    float bv0 = b1[n0 + (lane & 15)], bv1 = b1[n1 + (lane & 15)];
    for (int mt = 0; mt < 4; ++mt) {
      v8f c0 = {}, c1 = {};
#pragma unroll
      for (int kk = 0; kk < 8; ++kk) {
        v16h a  = load_a_lds(X, 256, mt * 16, kk * 32, lane);
        v16h m0 = load_b_glb(w1t, 256, n0, kk * 32, lane);
        v16h m1 = load_b_glb(w1t, 256, n1, kk * 32, lane);
        c0 = wmma16(a, m0, c0);
        c1 = wmma16(a, m1, c1);
      }
      int row = mt * 16 + ((lane >> 4) << 3);
      _Float16* d0 = A1 + row * 256 + n0 + (lane & 15);
      _Float16* d1 = A1 + row * 256 + n1 + (lane & 15);
#pragma unroll
      for (int j = 0; j < 8; ++j) {
        d0[j * 256] = (_Float16)fmaxf(c0[j] + bv0, 0.0f);
        d1[j * 256] = (_Float16)fmaxf(c1[j] + bv1, 0.0f);
      }
    }
  }
  __syncthreads();

  // ---- GEMM2: X = A1 @ W2 + b2 (no relu), reuse X buffer
  {
    float bv0 = b2[n0 + (lane & 15)], bv1 = b2[n1 + (lane & 15)];
    for (int mt = 0; mt < 4; ++mt) {
      v8f c0 = {}, c1 = {};
#pragma unroll
      for (int kk = 0; kk < 8; ++kk) {
        v16h a  = load_a_lds(A1, 256, mt * 16, kk * 32, lane);
        v16h m0 = load_b_glb(w2t, 256, n0, kk * 32, lane);
        v16h m1 = load_b_glb(w2t, 256, n1, kk * 32, lane);
        c0 = wmma16(a, m0, c0);
        c1 = wmma16(a, m1, c1);
      }
      int row = mt * 16 + ((lane >> 4) << 3);
      _Float16* d0 = X + row * 256 + n0 + (lane & 15);
      _Float16* d1 = X + row * 256 + n1 + (lane & 15);
#pragma unroll
      for (int j = 0; j < 8; ++j) {
        d0[j * 256] = (_Float16)(c0[j] + bv0);
        d1[j * 256] = (_Float16)(c1[j] + bv1);
      }
    }
  }
  __syncthreads();

  // ---- GEMM3: out[:,0:100] = X @ se_w + se_b   (N padded to 112, 7 n-tiles)
  if (w < 7) {
    int nb = w * 16;
    int n = nb + (lane & 15);
    for (int mt = 0; mt < 4; ++mt) {
      v8f c = {};
#pragma unroll
      for (int kk = 0; kk < 8; ++kk) {
        v16h a = load_a_lds(X, 256, mt * 16, kk * 32, lane);
        v16h m = load_b_glb(sewt, 256, nb, kk * 32, lane);
        c = wmma16(a, m, c);
      }
      if (n < 100) {
        float bv = seb[n];
        int row = rbase + mt * 16 + ((lane >> 4) << 3);
#pragma unroll
        for (int j = 0; j < 8; ++j)
          out[(size_t)(row + j) * OUTC + n] = c[j] + bv;
      }
    }
  }
}

// ---------------------------------------------------------------------------
// PointNet path: per object 128x3 -> 64 -> 128 -> 256, max over points,
// then 256->100 GEMV. One block per object, 8 waves.
// ---------------------------------------------------------------------------
__global__ __launch_bounds__(256) void pointnet_kernel(
    const float* __restrict__ pts,
    const float* __restrict__ pw1, const float* __restrict__ pb1,
    const _Float16* __restrict__ pw2t, const float* __restrict__ pb2,
    const _Float16* __restrict__ pw3t, const float* __restrict__ pb3,
    const _Float16* __restrict__ oewt, const float* __restrict__ oeb,
    float* __restrict__ out) {
  __shared__ float sP[128 * 3];
  __shared__ float sW1[192];
  __shared__ float sB1[64];
  __shared__ __align__(16) _Float16 H1[128 * 64];
  __shared__ __align__(16) _Float16 H2[128 * 128];
  __shared__ float sFeat[256];

  const int t = threadIdx.x, lane = t & 31, w = t >> 5;
  const int obj = blockIdx.x;
  const float* p0 = pts + (size_t)obj * (128 * 3);

  // ---- stage points + layer-1 weights
  if (t < 192)       sW1[t] = pw1[t];
  else if (t < 256)  sB1[t - 192] = pb1[t - 192];
  for (int i = t; i < 384; i += 256) sP[i] = p0[i];
  __syncthreads();

  // ---- layer 1 (K=3): VALU, output f16 H1 [128][64]
  for (int idx = t; idx < 128 * 64; idx += 256) {
    int p = idx >> 6, c = idx & 63;
    float v = sP[p * 3 + 0] * sW1[c] + sP[p * 3 + 1] * sW1[64 + c] +
              sP[p * 3 + 2] * sW1[128 + c] + sB1[c];
    H1[idx] = (_Float16)fmaxf(v, 0.0f);
  }
  __syncthreads();

  // ---- layer 2: H2 = relu(H1 @ W2 + b2)  (M=128,N=128,K=64); wave owns n-tile w
  {
    int nb = w * 16;
    float bv = pb2[nb + (lane & 15)];
    for (int mt = 0; mt < 8; ++mt) {
      v8f c = {};
#pragma unroll
      for (int kk = 0; kk < 2; ++kk) {
        v16h a = load_a_lds(H1, 64, mt * 16, kk * 32, lane);
        v16h m = load_b_glb(pw2t, 64, nb, kk * 32, lane);
        c = wmma16(a, m, c);
      }
      int row = mt * 16 + ((lane >> 4) << 3);
      _Float16* d = H2 + row * 128 + nb + (lane & 15);
#pragma unroll
      for (int j = 0; j < 8; ++j)
        d[j * 128] = (_Float16)fmaxf(c[j] + bv, 0.0f);
    }
  }
  __syncthreads();

  // ---- layer 3 + column max (M=128,N=256,K=128); wave owns n-tiles {w, w+8}
  {
    int na = w * 16, nb = (w + 8) * 16;
    float bv0 = pb3[na + (lane & 15)], bv1 = pb3[nb + (lane & 15)];
    float m0 = -3.402823466e38f, m1 = -3.402823466e38f;
    for (int mt = 0; mt < 8; ++mt) {
      v8f c0 = {}, c1 = {};
#pragma unroll
      for (int kk = 0; kk < 4; ++kk) {
        v16h a  = load_a_lds(H2, 128, mt * 16, kk * 32, lane);
        v16h w0 = load_b_glb(pw3t, 128, na, kk * 32, lane);
        v16h w1 = load_b_glb(pw3t, 128, nb, kk * 32, lane);
        c0 = wmma16(a, w0, c0);
        c1 = wmma16(a, w1, c1);
      }
#pragma unroll
      for (int j = 0; j < 8; ++j) {
        m0 = fmaxf(m0, c0[j] + bv0);
        m1 = fmaxf(m1, c1[j] + bv1);
      }
    }
    // lane L and L^16 hold the same output column -> merge halves
    m0 = fmaxf(m0, __shfl_xor(m0, 16, 32));
    m1 = fmaxf(m1, __shfl_xor(m1, 16, 32));
    if (lane < 16) {
      sFeat[na + lane] = m0;
      sFeat[nb + lane] = m1;
    }
  }
  __syncthreads();

  // ---- output embedding GEMV: out[:,100:200] = feat @ oe_w + oe_b
  if (t < 100) {
    const _Float16* wrow = oewt + (size_t)t * 256;
    float acc = 0.0f;
#pragma unroll 8
    for (int k = 0; k < 256; ++k) acc += sFeat[k] * (float)wrow[k];
    out[(size_t)obj * OUTC + 100 + t] = acc + oeb[t];
  }
}

// ---------------------------------------------------------------------------
extern "C" void kernel_launch(void* const* d_in, const int* in_sizes, int n_in,
                              void* d_out, int out_size, void* d_ws,
                              size_t ws_size, hipStream_t stream) {
  (void)in_sizes; (void)n_in; (void)out_size; (void)ws_size;

  const float* tot_pts = (const float*)d_in[0];
  const float* srcf    = (const float*)d_in[1];
  const float* reff    = (const float*)d_in[2];
  const float* sg_w1   = (const float*)d_in[6];
  const float* sg_b1   = (const float*)d_in[7];
  const float* sg_w2   = (const float*)d_in[8];
  const float* sg_b2   = (const float*)d_in[9];
  const float* se_w    = (const float*)d_in[10];
  const float* se_b    = (const float*)d_in[11];
  const float* p_w1    = (const float*)d_in[12];
  const float* p_b1    = (const float*)d_in[13];
  const float* p_w2    = (const float*)d_in[14];
  const float* p_b2    = (const float*)d_in[15];
  const float* p_w3    = (const float*)d_in[16];
  const float* p_b3    = (const float*)d_in[17];
  const float* oe_w    = (const float*)d_in[18];
  const float* oe_b    = (const float*)d_in[19];
  float* out = (float*)d_out;
  _Float16* ws = (_Float16*)d_ws;

  auto cvt = [&](const float* s, _Float16* d, int K, int N, int Nd) {
    int total = Nd * K;
    cvt_transpose<<<(total + 255) / 256, 256, 0, stream>>>(s, d, K, N, Nd);
  };
  cvt(sg_w1, ws + OFF_W1T,  256, 256, 256);
  cvt(sg_w2, ws + OFF_W2T,  256, 256, 256);
  cvt(se_w,  ws + OFF_SEWT, 256, 100, 112);
  cvt(p_w2,  ws + OFF_PW2T,  64, 128, 128);
  cvt(p_w3,  ws + OFF_PW3T, 128, 256, 256);
  cvt(oe_w,  ws + OFF_OEWT, 256, 100, 100);

  sg_kernel<<<256, 256, 0, stream>>>(srcf, reff,
      ws + OFF_W1T, ws + OFF_W2T, ws + OFF_SEWT,
      sg_b1, sg_b2, se_b, out);

  pointnet_kernel<<<16384, 256, 0, stream>>>(tot_pts,
      p_w1, p_b1, ws + OFF_PW2T, p_b2, ws + OFF_PW3T, p_b3,
      ws + OFF_OEWT, oe_b, out);
}
